// Recommender_67336497267221
// MI455X (gfx1250) — matrix operand order repaired and checked by reading, not compiled, not run on hardware
//
#include <hip/hip_runtime.h>
#include <hip/hip_bf16.h>

typedef __attribute__((ext_vector_type(16))) _Float16 v16h;
typedef __attribute__((ext_vector_type(8)))  float    v8f;

#define CH    64
#define NFAC  4
#define NMETA 8

// ---------------- KG edge aggregation: one wave32 per edge ----------------
// lane handles channels [2*lane, 2*lane+1] as float2.
__global__ void edge_agg_kernel(const float* __restrict__ entity_emb,
                                const int*   __restrict__ head,
                                const int*   __restrict__ tail,
                                const int*   __restrict__ edge_type,
                                const float* __restrict__ rel_w,
                                float*       __restrict__ entity_agg,
                                float*       __restrict__ cnt,
                                int n_edges) {
    int wave = threadIdx.x >> 5;
    int lane = threadIdx.x & 31;
    int e = blockIdx.x * 8 + wave;
    if (e >= n_edges) return;

    int h = head[e];
    int t = tail[e];
    int r = edge_type[e] - 1;

    const float2* eh = (const float2*)(entity_emb + (size_t)h * CH);
    const float2* et = (const float2*)(entity_emb + (size_t)t * CH);
    const float2* er = (const float2*)(rel_w      + (size_t)r * CH);

    float2 vh = eh[lane];
    float2 vr = er[lane];
    float dot = vh.x * vr.x + vh.y * vr.y;
    // full-wave butterfly reduction (wave32)
    #pragma unroll
    for (int m = 16; m >= 1; m >>= 1) dot += __shfl_xor(dot, m, 32);

    float att = 1.0f / (1.0f + __expf(-dot));

    float2 vt = et[lane];
    float nx = att * vt.x * vr.x;
    float ny = att * vt.y * vr.y;

    float* dst = entity_agg + (size_t)h * CH + lane * 2;
    atomicAdd(dst,     nx);
    atomicAdd(dst + 1, ny);
    if (lane == 0) atomicAdd(cnt + h, 1.0f);
}

// ---------------- scatter-mean finalize ----------------
__global__ void entity_finalize_kernel(float* __restrict__ eagg,
                                       const float* __restrict__ cnt,
                                       size_t n_elem) {
    size_t i = (size_t)blockIdx.x * blockDim.x + threadIdx.x;
    if (i >= n_elem) return;
    float c = cnt[i >> 6];
    eagg[i] = eagg[i] / fmaxf(c, 1.0f);
}

// ---------------- sparse interact_mat @ entity_emb: one wave32 per nnz ----------------
__global__ void sparse_agg_kernel(const float* __restrict__ entity_emb,
                                  const int*   __restrict__ mat_row,
                                  const int*   __restrict__ mat_col,
                                  const float* __restrict__ mat_val,
                                  float*       __restrict__ uagg,
                                  int nnz) {
    int wave = threadIdx.x >> 5;
    int lane = threadIdx.x & 31;
    int i = blockIdx.x * 8 + wave;
    if (i >= nnz) return;

    int   rr = mat_row[i];
    int   cc = mat_col[i];
    float v  = mat_val[i];

    const float2* ev = (const float2*)(entity_emb + (size_t)cc * CH);
    float2 e = ev[lane];

    float* dst = uagg + (size_t)rr * CH + lane * 2;
    atomicAdd(dst,     v * e.x);
    atomicAdd(dst + 1, v * e.y);
}

// ---------------- disen_weight = softmax(disen_att, -1) @ weight : tiny ----------------
__global__ void disen_kernel(const float* __restrict__ datt,
                             const float* __restrict__ weight,
                             float*       __restrict__ disen) {
    int c = threadIdx.x;  // 0..63
    for (int f = 0; f < NFAC; ++f) {
        float mx = -1e30f;
        for (int m = 0; m < NMETA; ++m) mx = fmaxf(mx, datt[f * NMETA + m]);
        float ex[NMETA]; float s = 0.0f;
        for (int m = 0; m < NMETA; ++m) { ex[m] = __expf(datt[f * NMETA + m] - mx); s += ex[m]; }
        float inv = 1.0f / s;
        float acc = 0.0f;
        for (int m = 0; m < NMETA; ++m) acc += (ex[m] * inv) * weight[m * CH + c];
        disen[f * CH + c] = acc;
    }
}

// ---------------- user score GEMM via v_wmma_f32_16x16x32_f16 + in-place scale ----------
// Each wave32 computes logits for a 16-user tile: D(16x16) = A(16x32)B(32x16)+C,
// two K-steps covering K=64. N is padded 4 -> 16 with zeros, staged once in LDS
// so the B-fragment build is branch-free (no predicated scalar global loads).
__global__ void user_score_kernel(const float* __restrict__ user_emb,
                                  const float* __restrict__ latent_emb,
                                  const float* __restrict__ disen,
                                  float*       __restrict__ uagg,
                                  int n_users) {
    __shared__ _Float16 lds_lat[16 * CH];  // zero-padded B operand: 16 rows x 64 K (f16)
    __shared__ float    lds_sc[128 * NFAC];

    // Cooperative fill: rows >= NFAC are zeros. Divergence here is harmless.
    for (int idx = threadIdx.x; idx < 16 * CH; idx += blockDim.x) {
        int row = idx >> 6;
        int k   = idx & 63;
        lds_lat[idx] = (row < NFAC) ? (_Float16)latent_emb[row * CH + k] : (_Float16)0.0f;
    }
    __syncthreads();

    int wave = threadIdx.x >> 5;
    int lane = threadIdx.x & 31;
    int tile = blockIdx.x * 8 + wave;      // 16 users per wave-tile
    int m  = lane & 15;
    int hi = lane >> 4;
    int u  = tile * 16 + m;
    int uc = (u < n_users) ? u : 0;        // clamp loads; no divergence around WMMA

    v8f c = {};
    #pragma unroll
    for (int kstep = 0; kstep < 2; ++kstep) {
        v16h a, b;
        // A fragment: 16x32 f16. Lane (m,hi): halves 0..7 -> K = hi*8 + j,
        //                                     halves 8..15 -> K = 16 + hi*8 + j.
        const float* arow = user_emb + (size_t)uc * CH + kstep * 32;
        #pragma unroll
        for (int j = 0; j < 8; ++j) a[j]     = (_Float16)arow[hi * 8 + j];
        #pragma unroll
        for (int j = 0; j < 8; ++j) a[8 + j] = (_Float16)arow[16 + hi * 8 + j];
        // B fragment: 32x16 f16. Lane n = lane&15, halves 0..7 -> K = hi*8 + j, halves 8..15 -> K = 16 + hi*8 + j.
        // Branch-free contiguous LDS reads from the zero-padded staging buffer.
        const _Float16* brow = &lds_lat[m * CH + kstep * 32 + hi * 8];
        #pragma unroll
        for (int j = 0; j < 8; ++j) { b[j] = brow[j]; b[8 + j] = brow[16 + j]; }
        c = __builtin_amdgcn_wmma_f32_16x16x32_f16(false, a, false, b,
                                                   (short)0, c, false, false);
    }

    // D layout: lane n = lane&15, element r -> row M = r + 8*(lane>>4).
    if (m < NFAC) {
        #pragma unroll
        for (int r = 0; r < 8; ++r) {
            int ml = hi * 8 + r;
            lds_sc[(wave * 16 + ml) * NFAC + m] = c[r];
        }
    }
    __syncthreads();

    // softmax over the 4 factors per local user
    if (threadIdx.x < 128) {
        float* p = &lds_sc[threadIdx.x * NFAC];
        float mx = fmaxf(fmaxf(p[0], p[1]), fmaxf(p[2], p[3]));
        float e0 = __expf(p[0] - mx), e1 = __expf(p[1] - mx);
        float e2 = __expf(p[2] - mx), e3 = __expf(p[3] - mx);
        float inv = 1.0f / (e0 + e1 + e2 + e3);
        p[0] = e0 * inv; p[1] = e1 * inv; p[2] = e2 * inv; p[3] = e3 * inv;
    }
    __syncthreads();

    // user_agg[u][ch] *= 1 + sum_f score_f * disen[f][ch]   (coalesced, in place)
    int base_user = blockIdx.x * 128;
    for (int idx = threadIdx.x; idx < 128 * CH; idx += blockDim.x) {
        int ul = idx >> 6;
        int ch = idx & 63;
        int uu = base_user + ul;
        if (uu < n_users) {
            const float* sc = &lds_sc[ul * NFAC];
            float coeff = sc[0] * disen[ch]          + sc[1] * disen[CH + ch]
                        + sc[2] * disen[2 * CH + ch] + sc[3] * disen[3 * CH + ch];
            size_t o = (size_t)uu * CH + ch;
            uagg[o] = uagg[o] * (1.0f + coeff);
        }
    }
}

extern "C" void kernel_launch(void* const* d_in, const int* in_sizes, int n_in,
                              void* d_out, int out_size, void* d_ws, size_t ws_size,
                              hipStream_t stream) {
    (void)n_in; (void)ws_size;
    const float* entity_emb = (const float*)d_in[0];
    const float* user_emb   = (const float*)d_in[1];
    const float* latent_emb = (const float*)d_in[2];
    const int*   head       = (const int*)  d_in[3];
    const int*   tail       = (const int*)  d_in[4];
    const int*   edge_type  = (const int*)  d_in[5];
    const int*   mat_row    = (const int*)  d_in[6];
    const int*   mat_col    = (const int*)  d_in[7];
    const float* mat_val    = (const float*)d_in[8];
    const float* rel_w      = (const float*)d_in[9];
    const float* weight     = (const float*)d_in[10];
    const float* datt       = (const float*)d_in[11];

    int n_entities = in_sizes[0] / CH;
    int n_users    = in_sizes[1] / CH;
    int n_edges    = in_sizes[3];
    int nnz        = in_sizes[6];

    float* entity_agg = (float*)d_out;
    float* user_agg   = entity_agg + (size_t)n_entities * CH;

    float* cnt   = (float*)d_ws;          // n_entities floats
    float* disen = cnt + n_entities;      // NFAC*CH floats

    hipMemsetAsync(d_out, 0, (size_t)out_size * sizeof(float), stream);
    hipMemsetAsync(cnt,   0, (size_t)n_entities * sizeof(float), stream);

    disen_kernel<<<1, CH, 0, stream>>>(datt, weight, disen);

    edge_agg_kernel<<<(n_edges + 7) / 8, 256, 0, stream>>>(
        entity_emb, head, tail, edge_type, rel_w, entity_agg, cnt, n_edges);

    size_t ne = (size_t)n_entities * CH;
    entity_finalize_kernel<<<(int)((ne + 255) / 256), 256, 0, stream>>>(
        entity_agg, cnt, ne);

    sparse_agg_kernel<<<(nnz + 7) / 8, 256, 0, stream>>>(
        entity_emb, mat_row, mat_col, mat_val, user_agg, nnz);

    user_score_kernel<<<(n_users + 127) / 128, 256, 0, stream>>>(
        user_emb, latent_emb, disen, user_agg, n_users);
}
